// DistributionEstimator_34102040330521
// MI455X (gfx1250) — matrix operand order repaired and checked by reading, not compiled, not run on hardware
//
#include <hip/hip_runtime.h>
#include <hip/hip_bf16.h>

// ---- CDNA5 WMMA vector types ----
typedef __attribute__((ext_vector_type(2)))  float  v2f;
typedef __attribute__((ext_vector_type(8)))  float  v8f;
typedef __attribute__((ext_vector_type(16))) __bf16 v16bf;

static constexpr int NF   = 4096;   // features rows
static constexpr int NM   = 8192;   // memory rows
static constexpr int DIMS = 1024;   // feature dim
static constexpr int NEWTON_ITERS = 18;

// ---- CDNA5 async global->LDS copy (per-lane 16B, tracked by ASYNCcnt) ----
// LDS address from one VGPR, global address from a VGPR pair. No VGPR staging.
__device__ __forceinline__ void async_load_b128(unsigned lds_addr, const void* gaddr) {
  asm volatile("global_load_async_to_lds_b128 %0, %1, off"
               :: "v"(lds_addr), "v"(gaddr)
               : "memory");
}
__device__ __forceinline__ void wait_async0() {
  asm volatile("s_wait_asynccnt 0x0" ::: "memory");
}
__device__ __forceinline__ unsigned lds_addr_of(const void* p) {
  // Generic pointers to LDS carry the LDS byte offset in their low 32 bits.
  return (unsigned)(size_t)p;
}

// XOR-butterfly add-reduce across the 16 lanes of each half-wave using
// ds_swizzle_b32 (group-of-32 mode: offset = xor_mask<<10 | and_mask).
__device__ inline float halfwave_sum16(float v) {
  v += __int_as_float(__builtin_amdgcn_ds_swizzle(__float_as_int(v), (8 << 10) | 0x1f));
  v += __int_as_float(__builtin_amdgcn_ds_swizzle(__float_as_int(v), (4 << 10) | 0x1f));
  v += __int_as_float(__builtin_amdgcn_ds_swizzle(__float_as_int(v), (2 << 10) | 0x1f));
  v += __int_as_float(__builtin_amdgcn_ds_swizzle(__float_as_int(v), (1 << 10) | 0x1f));
  return v;
}

// =====================================================================
// Small utility kernels
// =====================================================================
__global__ __launch_bounds__(256) void zero_f32(float* p, int n) {
  int i = blockIdx.x * 256 + threadIdx.x;
  if (i < n) p[i] = 0.f;
}

__global__ __launch_bounds__(256) void f32_to_bf16(const float* __restrict__ src,
                                                   __bf16* __restrict__ dst, int n) {
  int i = blockIdx.x * 256 + threadIdx.x;
  if (i < n) dst[i] = (__bf16)src[i];
}

// Per-row: squared L2 norm (of the RAW row, used by the RBF kernel) and the
// L2-normalized row (torch F.normalize semantics, eps=1e-12).
__global__ __launch_bounds__(256) void rownorm_normalize(const float* __restrict__ src,
                                                         float* __restrict__ dst,
                                                         float* __restrict__ rn, int Dd) {
  int row = blockIdx.x;
  const float* s = src + (size_t)row * Dd;
  float part = 0.f;
  for (int j = threadIdx.x; j < Dd; j += 256) { float v = s[j]; part += v * v; }
  for (int m = 16; m > 0; m >>= 1) part += __shfl_xor(part, m, 32);
  __shared__ float wsum[8];
  __shared__ float s_tot;
  if ((threadIdx.x & 31) == 0) wsum[threadIdx.x >> 5] = part;
  __syncthreads();
  if (threadIdx.x == 0) {
    float t = 0.f;
    for (int w = 0; w < 8; ++w) t += wsum[w];
    s_tot = t;
    rn[row] = t;
  }
  __syncthreads();
  float inv = 1.f / fmaxf(sqrtf(s_tot), 1e-12f);
  float* d = dst + (size_t)row * Dd;
  for (int j = threadIdx.x; j < Dd; j += 256) d[j] = s[j] * inv;
}

__global__ __launch_bounds__(256) void colmean(const float* __restrict__ src,
                                               float* __restrict__ mean, int N, int Dd) {
  int j = blockIdx.x * 256 + threadIdx.x;
  if (j >= Dd) return;
  float s = 0.f;
  for (int i = 0; i < N; ++i) s += src[(size_t)i * Dd + j];
  mean[j] = s / (float)N;
}

// G currently holds M^T M; turn it into (centered^T centered)/(N-1) + eps*I
// using  centered^T centered = M^T M - N * mean mean^T.
__global__ __launch_bounds__(256) void cov_finalize(float* __restrict__ G,
                                                    const float* __restrict__ mean,
                                                    int N, int Dd) {
  int idx = blockIdx.x * 256 + threadIdx.x;
  if (idx >= Dd * Dd) return;
  int i = idx / Dd, j = idx % Dd;
  float g = (G[idx] - (float)N * mean[i] * mean[j]) / (float)(N - 1);
  if (i == j) g += 1e-6f;
  G[idx] = g;
}

__global__ __launch_bounds__(256) void rowabssum(const float* __restrict__ A,
                                                 float* __restrict__ out, int Dd) {
  int row = blockIdx.x;
  float part = 0.f;
  for (int j = threadIdx.x; j < Dd; j += 256) part += fabsf(A[(size_t)row * Dd + j]);
  for (int m = 16; m > 0; m >>= 1) part += __shfl_xor(part, m, 32);
  __shared__ float wsum[8];
  if ((threadIdx.x & 31) == 0) wsum[threadIdx.x >> 5] = part;
  __syncthreads();
  if (threadIdx.x == 0) {
    float t = 0.f;
    for (int w = 0; w < 8; ++w) t += wsum[w];
    out[row] = t;
  }
}

__global__ __launch_bounds__(256) void reduce_max1(const float* __restrict__ in, int n,
                                                   float* __restrict__ out) {
  float part = 0.f;  // inputs are non-negative
  for (int j = threadIdx.x; j < n; j += 256) part = fmaxf(part, in[j]);
  for (int m = 16; m > 0; m >>= 1) part = fmaxf(part, __shfl_xor(part, m, 32));
  __shared__ float wsum[8];
  if ((threadIdx.x & 31) == 0) wsum[threadIdx.x >> 5] = part;
  __syncthreads();
  if (threadIdx.x == 0) {
    float t = 0.f;
    for (int w = 0; w < 8; ++w) t = fmaxf(t, wsum[w]);
    out[0] = t;
  }
}

__global__ __launch_bounds__(256) void reduce_sum1(const float* __restrict__ in, int n,
                                                   float* __restrict__ out) {
  float part = 0.f;
  for (int j = threadIdx.x; j < n; j += 256) part += in[j];
  for (int m = 16; m > 0; m >>= 1) part += __shfl_xor(part, m, 32);
  __shared__ float wsum[8];
  if ((threadIdx.x & 31) == 0) wsum[threadIdx.x >> 5] = part;
  __syncthreads();
  if (threadIdx.x == 0) {
    float t = 0.f;
    for (int w = 0; w < 8; ++w) t += wsum[w];
    out[0] = t;
  }
}

// Newton-Schulz init: X0 = A / (||A||_1 * ||A||_inf); A symmetric -> norm1^2.
__global__ __launch_bounds__(256) void newton_init(float* __restrict__ X,
                                                   const float* __restrict__ G,
                                                   const float* __restrict__ norm1, int n) {
  int i = blockIdx.x * 256 + threadIdx.x;
  if (i < n) {
    float s = norm1[0];
    X[i] = G[i] / (s * s);
  }
}

// T := 2I - T
__global__ __launch_bounds__(256) void newton_rhs(float* __restrict__ T, int Dd) {
  int idx = blockIdx.x * 256 + threadIdx.x;
  if (idx >= Dd * Dd) return;
  int i = idx / Dd, j = idx % Dd;
  T[idx] = (i == j ? 2.f : 0.f) - T[idx];
}

__global__ __launch_bounds__(256) void center_kernel(const float* __restrict__ src,
                                                     const float* __restrict__ mean,
                                                     float* __restrict__ dst, int total, int Dd) {
  int idx = blockIdx.x * 256 + threadIdx.x;
  if (idx >= total) return;
  dst[idx] = src[idx] - mean[idx % Dd];
}

// out[row] = sum_j U[row][j] * CF[row][j]
__global__ __launch_bounds__(256) void rowdot(const float* __restrict__ U,
                                              const float* __restrict__ CF,
                                              float* __restrict__ out, int Dd) {
  int row = blockIdx.x;
  size_t base = (size_t)row * Dd;
  float part = 0.f;
  for (int j = threadIdx.x; j < Dd; j += 256) part += U[base + j] * CF[base + j];
  for (int m = 16; m > 0; m >>= 1) part += __shfl_xor(part, m, 32);
  __shared__ float wsum[8];
  if ((threadIdx.x & 31) == 0) wsum[threadIdx.x >> 5] = part;
  __syncthreads();
  if (threadIdx.x == 0) {
    float t = 0.f;
    for (int w = 0; w < 8; ++w) t += wsum[w];
    out[row] = t;
  }
}

__global__ __launch_bounds__(256) void combine_kernel(float* __restrict__ out,
                                                      const float* __restrict__ m,
                                                      const float* __restrict__ mpp,
                                                      const float* __restrict__ kxx,
                                                      const float* __restrict__ kxy,
                                                      const float* __restrict__ kyy_total, int n) {
  int i = blockIdx.x * 256 + threadIdx.x;
  if (i >= n) return;
  float kyy = kyy_total[0] / ((float)NM * (float)NM);
  float mmd = kxx[i] / (float)NF + kyy - 2.f * kxy[i] / (float)NM;
  out[i] = 0.5f * m[i] + 0.3f * mpp[i] + 0.2f * mmd;
}

// =====================================================================
// f32 WMMA GEMM (V_WMMA_F32_16X16X4_F32), 64x64 C tile, 8 waves/block,
// double-buffered LDS staging. B tile (and NN A tile) use async
// global->LDS b128 copies; the TN A tile needs a transposed scatter so it
// keeps VGPR staging.
// TRANSA=false: C[M,N] = A[M,K] @ B[K,N]  (all row-major)
// TRANSA=true : C[M,N] = A[K,M]^T @ B[K,N]
// All of M, N divisible by 64 and K by 16 in this problem (no edge paths).
// =====================================================================
template <bool TRANSA>
__global__ __launch_bounds__(256) void gemm_f32_wmma(float* __restrict__ C,
                                                     const float* __restrict__ A,
                                                     const float* __restrict__ B,
                                                     int M, int N, int K) {
  __shared__ __align__(16) float As[2][64][20];  // [m][k], 80B row stride
  __shared__ __align__(16) float Bs[2][16][64];  // [k][n], 256B row stride
  const int row0 = blockIdx.x * 64, col0 = blockIdx.y * 64;
  const int tid = threadIdx.x;
  const int lane = tid & 31, wave = tid >> 5;
  const int wm = wave & 3;   // which 16-row strip
  const int wn = wave >> 2;  // which 32-col half
  const int m16 = lane & 15; // M index (A/D) and N index (B/D) within fragment
  const int kh = lane >> 4;  // lane-half selects K pair (A/B f32 layout)

  // Per-thread staging coordinates.
  const int a_mm = tid >> 2, a_kk4 = (tid & 3) * 4;       // !TRANSA: one b128
  const int t_kk = tid >> 4, t_m4 = (tid & 15) * 4;       // TRANSA: contiguous along M
  const int b_kk = tid >> 4, b_n4 = (tid & 15) * 4;       // B: one b128

  v8f acc0 = {};
  v8f acc1 = {};
  float4 va;

  // Prologue: stage k0 = 0 into buffer 0.
  if (!TRANSA) {
    async_load_b128(lds_addr_of(&As[0][a_mm][a_kk4]),
                    A + (size_t)(row0 + a_mm) * K + a_kk4);
  } else {
    va = *(const float4*)(A + (size_t)t_kk * M + (row0 + t_m4));
    As[0][t_m4 + 0][t_kk] = va.x;
    As[0][t_m4 + 1][t_kk] = va.y;
    As[0][t_m4 + 2][t_kk] = va.z;
    As[0][t_m4 + 3][t_kk] = va.w;
  }
  async_load_b128(lds_addr_of(&Bs[0][b_kk][b_n4]),
                  B + (size_t)b_kk * N + (col0 + b_n4));
  wait_async0();
  __syncthreads();

  const int nstages = K >> 4;
  for (int s = 0; s < nstages; ++s) {
    const bool has_next = (s + 1) < nstages;
    const int buf = s & 1, nb = buf ^ 1;
    if (has_next) {
      int k0 = (s + 1) << 4;
      if (!TRANSA) {
        async_load_b128(lds_addr_of(&As[nb][a_mm][a_kk4]),
                        A + (size_t)(row0 + a_mm) * K + (k0 + a_kk4));
      } else {
        va = *(const float4*)(A + (size_t)(k0 + t_kk) * M + (row0 + t_m4));
      }
      async_load_b128(lds_addr_of(&Bs[nb][b_kk][b_n4]),
                      B + (size_t)(k0 + b_kk) * N + (col0 + b_n4));
    }
#pragma unroll
    for (int kk = 0; kk < 16; kk += 4) {
      // A fragment 16x4: lanes 0-15 hold K = kk+{0,1}, lanes 16-31 K = kk+{2,3}
      v2f a;
      a.x = As[buf][wm * 16 + m16][kk + 2 * kh];
      a.y = As[buf][wm * 16 + m16][kk + 2 * kh + 1];
      v2f b0, b1;
      b0.x = Bs[buf][kk + 2 * kh][wn * 32 + m16];
      b0.y = Bs[buf][kk + 2 * kh + 1][wn * 32 + m16];
      b1.x = Bs[buf][kk + 2 * kh][wn * 32 + 16 + m16];
      b1.y = Bs[buf][kk + 2 * kh + 1][wn * 32 + 16 + m16];
      acc0 = __builtin_amdgcn_wmma_f32_16x16x4_f32(false, a, false, b0, (short)0, acc0, false, false);
      acc1 = __builtin_amdgcn_wmma_f32_16x16x4_f32(false, a, false, b1, (short)0, acc1, false, false);
    }
    if (has_next) {
      if (TRANSA) {
        As[nb][t_m4 + 0][t_kk] = va.x;
        As[nb][t_m4 + 1][t_kk] = va.y;
        As[nb][t_m4 + 2][t_kk] = va.z;
        As[nb][t_m4 + 3][t_kk] = va.w;
      }
      wait_async0();
      __syncthreads();
    }
  }

#pragma unroll
  for (int v = 0; v < 8; ++v) {
    int r = row0 + wm * 16 + v + 8 * kh;  // C/D layout: VGPR v -> M=v (lanes<16) / v+8
    int c = col0 + wn * 32 + m16;
    C[(size_t)r * N + c] = acc0[v];
    C[(size_t)r * N + c + 16] = acc1[v];
  }
}

// =====================================================================
// Fused RBF row-sum via bf16 WMMA (V_WMMA_F32_16X16X32_BF16):
//   out_rows[i] += sum_j exp(-0.5 * max(rnx[i] + rny[j] - 2 <X_i, Y_j>, 0))
// X:[NX,D] Y:[NY,D] bf16 row-major; NX divisible by 128, NY by 64, D by 32.
// Block tile: 128 (X rows) x 64 (Y rows); 8 waves, each wave owns a 16-row
// strip and 4 accumulators spanning all 64 columns (A-fragment reuse x4).
// Double-buffered; all staging via async global->LDS b128 (no VGPR staging),
// overlapped with the WMMAs of the current stage; one barrier per K-stage.
// =====================================================================
__global__ __launch_bounds__(256) void gauss_rowsum_wmma(const __bf16* __restrict__ X,
                                                         const __bf16* __restrict__ Y,
                                                         const float* __restrict__ rnx,
                                                         const float* __restrict__ rny,
                                                         float* __restrict__ out_rows) {
  __shared__ __align__(16) __bf16 Xs[2][128][40];  // 32 K + 8 pad -> 80B row
  __shared__ __align__(16) __bf16 Ys[2][64][40];
  __shared__ float rowsum[128];
  const int i0 = blockIdx.x * 128, j0 = blockIdx.y * 64;
  const int tid = threadIdx.x;
  const int lane = tid & 31, wave = tid >> 5;  // wave = 16-row strip (0..7)
  const int n16 = lane & 15, kh = lane >> 4;

  // Staging: X tile = 128x32 bf16 = 512 x b128 (2/thread at rows sr, sr+64),
  // Y tile = 64x32 bf16 = 256 x b128 (1/thread). se = element offset in row.
  const int sr = tid >> 2, se = (tid & 3) * 8;

  if (tid < 128) rowsum[tid] = 0.f;

  v8f acc[4] = {{}, {}, {}, {}};

  // Prologue: stage k0 = 0 into buffer 0.
  async_load_b128(lds_addr_of(&Xs[0][sr][se]),      X + (size_t)(i0 + sr) * DIMS + se);
  async_load_b128(lds_addr_of(&Xs[0][64 + sr][se]), X + (size_t)(i0 + 64 + sr) * DIMS + se);
  async_load_b128(lds_addr_of(&Ys[0][sr][se]),      Y + (size_t)(j0 + sr) * DIMS + se);
  wait_async0();
  __syncthreads();

  const int nstages = DIMS / 32;
  for (int s = 0; s < nstages; ++s) {
    const bool has_next = (s + 1) < nstages;
    const int buf = s & 1, nb = buf ^ 1;
    if (has_next) {
      int k0 = (s + 1) * 32;
      async_load_b128(lds_addr_of(&Xs[nb][sr][se]),
                      X + (size_t)(i0 + sr) * DIMS + k0 + se);
      async_load_b128(lds_addr_of(&Xs[nb][64 + sr][se]),
                      X + (size_t)(i0 + 64 + sr) * DIMS + k0 + se);
      async_load_b128(lds_addr_of(&Ys[nb][sr][se]),
                      Y + (size_t)(j0 + sr) * DIMS + k0 + se);
    }

    // A fragment (16x32 bf16): lanes<16 hold K 0-7,16-23; lanes>=16 K 8-15,24-31
    v16bf a;
    int arow = wave * 16 + n16;
#pragma unroll
    for (int e = 0; e < 16; ++e) {
      int ka = e + (e >= 8 ? 8 : 0) + 8 * kh;
      a[e] = Xs[buf][arow][ka];
    }
    // B fragments (32x16): lanes<16 hold K 0-7,16-23; lanes>=16 K 8-15,24-31; B[k][n]=Y[j0+n][k]
#pragma unroll
    for (int c = 0; c < 4; ++c) {
      v16bf b;
#pragma unroll
      for (int e = 0; e < 16; ++e) b[e] = Ys[buf][c * 16 + n16][e + (e >= 8 ? 8 : 0) + 8 * kh];
      acc[c] = __builtin_amdgcn_wmma_f32_16x16x32_bf16(false, a, false, b, (short)0,
                                                       acc[c], false, false);
    }
    if (has_next) {
      wait_async0();
      __syncthreads();
    }
  }

  // Epilogue: exp(-d/2) and row reduction (N = lane&15 within each half-wave).
#pragma unroll
  for (int c = 0; c < 4; ++c) {
    int colg = j0 + c * 16 + n16;
    float ry2 = rny[colg];
#pragma unroll
    for (int v = 0; v < 8; ++v) {
      int rloc = wave * 16 + v + 8 * kh;
      float dist = fmaxf(rnx[i0 + rloc] + ry2 - 2.f * acc[c][v], 0.f);
      float val = halfwave_sum16(__expf(-0.5f * dist));
      if (n16 == 0) atomicAdd(&rowsum[rloc], val);  // ds_add_f32
    }
  }
  __syncthreads();
  if (tid < 128) atomicAdd(&out_rows[i0 + tid], rowsum[tid]);
}

// =====================================================================
// Host-side orchestration
// =====================================================================
extern "C" void kernel_launch(void* const* d_in, const int* in_sizes, int n_in,
                              void* d_out, int out_size, void* d_ws, size_t ws_size,
                              hipStream_t stream) {
  const float* features = (const float*)d_in[0];  // [4096,1024]
  const float* memory   = (const float*)d_in[1];  // [8192,1024]
  float* out = (float*)d_out;                     // [4096]

  // Workspace carve-out (~130 MB total).
  char* base = (char*)d_ws;
  size_t off = 0;
  auto alloc = [&](size_t bytes) -> char* {
    char* p = base + off;
    off += (bytes + 255) & ~(size_t)255;
    return p;
  };
  float*  Fn       = (float*)alloc((size_t)NF * DIMS * 4);   // normalized features
  float*  Mn       = (float*)alloc((size_t)NM * DIMS * 4);   // normalized memory
  float*  CF       = (float*)alloc((size_t)NF * DIMS * 4);   // centered features (per path)
  float*  U        = (float*)alloc((size_t)NF * DIMS * 4);   // CF @ cov_inv
  __bf16* Fbf      = (__bf16*)alloc((size_t)NF * DIMS * 2);
  __bf16* Mbf      = (__bf16*)alloc((size_t)NM * DIMS * 2);
  float*  G        = (float*)alloc((size_t)DIMS * DIMS * 4); // Gram -> covariance
  float*  Xa       = (float*)alloc((size_t)DIMS * DIMS * 4); // Newton iterate
  float*  Xb       = (float*)alloc((size_t)DIMS * DIMS * 4);
  float*  T0       = (float*)alloc((size_t)DIMS * DIMS * 4);
  float*  mean_m   = (float*)alloc(DIMS * 4);
  float*  mean_p   = (float*)alloc(DIMS * 4);
  float*  rn_f     = (float*)alloc(NF * 4);
  float*  rn_m     = (float*)alloc(NM * 4);
  float*  m_out    = (float*)alloc(NF * 4);
  float*  mpp_out  = (float*)alloc(NF * 4);
  float*  kxx      = (float*)alloc(NF * 4);
  float*  kxy      = (float*)alloc(NF * 4);
  float*  kyy_rows = (float*)alloc(NM * 4);
  float*  tmpD     = (float*)alloc(DIMS * 4);
  float*  scal     = (float*)alloc(256);  // [0]=norm1, [1]=kyy total
  (void)ws_size; (void)in_sizes; (void)n_in; (void)out_size;

  // ---- normalization + raw squared row norms + bf16 copies ----
  rownorm_normalize<<<NF, 256, 0, stream>>>(features, Fn, rn_f, DIMS);
  rownorm_normalize<<<NM, 256, 0, stream>>>(memory, Mn, rn_m, DIMS);
  f32_to_bf16<<<(NF * DIMS + 255) / 256, 256, 0, stream>>>(features, Fbf, NF * DIMS);
  f32_to_bf16<<<(NM * DIMS + 255) / 256, 256, 0, stream>>>(memory, Mbf, NM * DIMS);
  colmean<<<(DIMS + 255) / 256, 256, 0, stream>>>(memory, mean_m, NM, DIMS);
  colmean<<<(DIMS + 255) / 256, 256, 0, stream>>>(Mn, mean_p, NM, DIMS);

  // ---- one Mahalanobis path: bank -> cov -> Newton inverse -> score ----
  dim3 gDD(DIMS / 64, DIMS / 64);
  dim3 gFD(NF / 64, DIMS / 64);
  auto run_path = [&](const float* bank, const float* feat, const float* mean,
                      float* result) {
    gemm_f32_wmma<true><<<gDD, 256, 0, stream>>>(G, bank, bank, DIMS, DIMS, NM);
    cov_finalize<<<(DIMS * DIMS + 255) / 256, 256, 0, stream>>>(G, mean, NM, DIMS);
    rowabssum<<<DIMS, 256, 0, stream>>>(G, tmpD, DIMS);
    reduce_max1<<<1, 256, 0, stream>>>(tmpD, DIMS, scal);
    newton_init<<<(DIMS * DIMS + 255) / 256, 256, 0, stream>>>(Xa, G, scal, DIMS * DIMS);
    float* Xc = Xa;
    float* Xt = Xb;
    for (int it = 0; it < NEWTON_ITERS; ++it) {
      gemm_f32_wmma<false><<<gDD, 256, 0, stream>>>(T0, G, Xc, DIMS, DIMS, DIMS);
      newton_rhs<<<(DIMS * DIMS + 255) / 256, 256, 0, stream>>>(T0, DIMS);
      gemm_f32_wmma<false><<<gDD, 256, 0, stream>>>(Xt, Xc, T0, DIMS, DIMS, DIMS);
      float* sw = Xc; Xc = Xt; Xt = sw;
    }
    center_kernel<<<(NF * DIMS + 255) / 256, 256, 0, stream>>>(feat, mean, CF, NF * DIMS, DIMS);
    gemm_f32_wmma<false><<<gFD, 256, 0, stream>>>(U, CF, Xc, NF, DIMS, DIMS);
    rowdot<<<NF, 256, 0, stream>>>(U, CF, result, DIMS);
  };
  run_path(memory, features, mean_m, m_out);  // raw path
  run_path(Mn, Fn, mean_p, mpp_out);          // normalized ("++") path

  // ---- MMD terms (fused bf16 WMMA Gram + exp + row-mean) ----
  zero_f32<<<(NF + 255) / 256, 256, 0, stream>>>(kxx, NF);
  zero_f32<<<(NF + 255) / 256, 256, 0, stream>>>(kxy, NF);
  zero_f32<<<(NM + 255) / 256, 256, 0, stream>>>(kyy_rows, NM);
  gauss_rowsum_wmma<<<dim3(NF / 128, NF / 64), 256, 0, stream>>>(Fbf, Fbf, rn_f, rn_f, kxx);
  gauss_rowsum_wmma<<<dim3(NF / 128, NM / 64), 256, 0, stream>>>(Fbf, Mbf, rn_f, rn_m, kxy);
  gauss_rowsum_wmma<<<dim3(NM / 128, NM / 64), 256, 0, stream>>>(Mbf, Mbf, rn_m, rn_m, kyy_rows);
  reduce_sum1<<<1, 256, 0, stream>>>(kyy_rows, NM, scal + 1);

  // ---- final combine ----
  combine_kernel<<<(NF + 255) / 256, 256, 0, stream>>>(out, m_out, mpp_out, kxx, kxy,
                                                       scal + 1, NF);
}
